// GraphAttConv_17308718203021
// MI455X (gfx1250) — matrix operand order repaired and checked by reading, not compiled, not run on hardware
//
#include <hip/hip_runtime.h>
#include <hip/hip_bf16.h>
#include <math.h>

typedef __attribute__((ext_vector_type(2)))  float  v2f;
typedef __attribute__((ext_vector_type(8)))  float  v8f;
typedef __attribute__((ext_vector_type(16))) __bf16 v16bf;
typedef __attribute__((ext_vector_type(4)))  unsigned int u32x4;
typedef __attribute__((ext_vector_type(4)))  int    i32x4;
typedef __attribute__((ext_vector_type(8)))  int    i32x8;

#define ALPHA_  0.2f
#define NEGINF_ (-9.0e15f)
#define B_    8
#define N_    2048
#define FIN_  128
#define FOUT_ 64

#if defined(__gfx1250__) && __has_builtin(__builtin_amdgcn_tensor_load_to_lds) && \
    __has_builtin(__builtin_amdgcn_s_wait_tensorcnt)
#define HAVE_TDM 1
#else
#define HAVE_TDM 0
#endif

// LDS tile geometry for TDM-staged adj tiles: 16 rows x 32 fp32 cols,
// +4 dwords pad per row (TDM pad_interval=32dw, pad_amount=4dw) ->
// row stride 36 dwords: 16B-aligned rows, banks spread.
#define TROW_ 36
#define TSZ_  (16 * TROW_)   // 576 dwords per tile buffer

#if HAVE_TDM
// Issue one TDM tile load: 16 rows x 32 fp32 cols of adj -> LDS (padded rows).
__device__ __forceinline__ void tdm_load_tile(unsigned ldsOff, const float* gsrc)
{
    const unsigned long long ga = (unsigned long long)(const void*)gsrc;
    u32x4 g0;
    g0[0] = 1u;                                              // count=1 (valid D#)
    g0[1] = ldsOff;                                          // lds_addr
    g0[2] = (unsigned)(ga & 0xFFFFFFFFull);                  // global_addr[31:0]
    g0[3] = (unsigned)((ga >> 32) & 0x01FFFFFFull) | (2u << 30); // addr[56:32] | type=2
    i32x8 g1;
    // data_size=4B(2), pad_enable, pad_interval=32dw(4), pad_amount=4dw(3)
    g1[0] = (2 << 16) | (1 << 20) | (4 << 22) | (3 << 25);
    g1[1] = (int)((unsigned)(N_ & 0xFFFF) << 16);            // tensor_dim0[15:0]=2048
    g1[2] = (int)((unsigned)(N_ & 0xFFFF) << 16);            // dim0[31:16]=0 | tensor_dim1[15:0]=2048
    g1[3] = (32 << 16);                                      // dim1[31:16]=0 | tile_dim0=32
    g1[4] = 16;                                              // tile_dim1=16, tile_dim2=0
    g1[5] = N_;                                              // tensor_dim0_stride=2048
    g1[6] = 0;
    g1[7] = 0;
    const i32x4 gz = {0, 0, 0, 0};
#if defined(__clang_major__) && __clang_major__ >= 23
    const i32x8 gz8 = {0, 0, 0, 0, 0, 0, 0, 0};
    __builtin_amdgcn_tensor_load_to_lds(g0, g1, gz, gz, gz8, 0);
#else
    __builtin_amdgcn_tensor_load_to_lds(g0, g1, gz, gz, 0);
#endif
}
#endif

// -------------------------------------------------------------------------
// Kernel 2 (emitted FIRST so the disasm snippet shows it): fused masked
// softmax (online) + attention@hidden via bf16 WMMA.  adj tiles streamed by
// the Tensor Data Mover into double-buffered, bank-padded LDS.
// -------------------------------------------------------------------------
__global__ __launch_bounds__(256) void gat_attn_kernel(
    const float* __restrict__ adj, const __bf16* __restrict__ hiddenT,
    const float* __restrict__ s1, const float* __restrict__ s2,
    float* __restrict__ out)
{
    // TDM staging buffers (loop phase) overlaid with combine buffer (epilogue)
    __shared__ __align__(16) union SMem {
        float adjbuf[8 * 2 * TSZ_];      // 36,864 B
        float accL[8][16][64];           // 32,768 B
    } smem;
    __shared__ float mL[8][16];
    __shared__ float lL[8][16];
    __shared__ float wfac[8][16];
    __shared__ float invL[16];

    const int tid  = threadIdx.x;
    const int lane = tid & 31;
    const int wv   = __builtin_amdgcn_readfirstlane(tid >> 5);   // uniform wave id
    const int half = lane >> 4, rl = lane & 15;
    const int b  = blockIdx.x >> 7;
    const int r0 = (blockIdx.x & 127) * 16;

    const float  s1r  = s1[b * N_ + r0 + rl];
    const float* s2b  = s2 + b * N_;
    const __bf16* hTb = hiddenT + (size_t)b * FOUT_ * N_;
    const float* adjBase = adj + ((size_t)b * N_ + r0) * N_;     // 16-row slab
#if !HAVE_TDM
    const float* arow = adjBase + (size_t)rl * N_;
#endif

    float m = NEGINF_, l = 0.f;
    v8f acc0 = {}, acc1 = {}, acc2 = {}, acc3 = {};

    const int cbase = wv * 256;
#if HAVE_TDM
    float* abuf0 = smem.adjbuf + wv * 2 * TSZ_;
    float* abuf1 = abuf0 + TSZ_;
    const unsigned off0 = (unsigned)(uintptr_t)(void*)abuf0;
    const unsigned off1 = (unsigned)(uintptr_t)(void*)abuf1;
    tdm_load_tile(off0, adjBase + cbase);                        // prologue
#endif

    for (int it = 0; it < 8; ++it) {
        const int c0 = cbase + it * 32;
        float4 aA, aB, aC, aD;
#if HAVE_TDM
        if (it < 7) {
            tdm_load_tile(((it + 1) & 1) ? off1 : off0, adjBase + c0 + 32);
            __builtin_amdgcn_s_wait_tensorcnt(1);    // current tile resident
        } else {
            __builtin_amdgcn_s_wait_tensorcnt(0);
        }
        asm volatile("" ::: "memory");
        const float* tb = (it & 1) ? abuf1 : abuf0;
        const float* rowp = tb + rl * TROW_ + half * 16;
        aA = *(const float4*)(rowp);
        aB = *(const float4*)(rowp + 4);
        aC = *(const float4*)(rowp + 8);
        aD = *(const float4*)(rowp + 12);
#else
        const int g1i = c0 + half * 16;
        aA = *(const float4*)(arow + g1i);
        aB = *(const float4*)(arow + g1i + 4);
        aC = *(const float4*)(arow + g1i + 8);
        aD = *(const float4*)(arow + g1i + 12);
#endif
        const int gA = c0 + half * 16;
        const float4 sA = *(const float4*)(s2b + gA);
        const float4 sB = *(const float4*)(s2b + gA + 4);
        const float4 sC = *(const float4*)(s2b + gA + 8);
        const float4 sD = *(const float4*)(s2b + gA + 12);

        float sc[16];
        auto SCORE = [&](float av, float sv) -> float {
            float e = s1r + sv;
            e = (e >= 0.f) ? e : ALPHA_ * e;
            return (av > 0.f) ? e * av : NEGINF_;
        };
        sc[0]  = SCORE(aA.x, sA.x); sc[1]  = SCORE(aA.y, sA.y);
        sc[2]  = SCORE(aA.z, sA.z); sc[3]  = SCORE(aA.w, sA.w);
        sc[4]  = SCORE(aB.x, sB.x); sc[5]  = SCORE(aB.y, sB.y);
        sc[6]  = SCORE(aB.z, sB.z); sc[7]  = SCORE(aB.w, sB.w);
        sc[8]  = SCORE(aC.x, sC.x); sc[9]  = SCORE(aC.y, sC.y);
        sc[10] = SCORE(aC.z, sC.z); sc[11] = SCORE(aC.w, sC.w);
        sc[12] = SCORE(aD.x, sD.x); sc[13] = SCORE(aD.y, sD.y);
        sc[14] = SCORE(aD.z, sD.z); sc[15] = SCORE(aD.w, sD.w);

        float tmax = sc[0];
        #pragma unroll
        for (int i = 1; i < 16; ++i) tmax = fmaxf(tmax, sc[i]);
        tmax = fmaxf(tmax, __shfl_xor(tmax, 16, 32));
        const float m_new = fmaxf(m, tmax);
        const float scale = __expf(m - m_new);

        v16bf pa;
        float psum = 0.f;
        #pragma unroll
        for (int i = 0; i < 16; ++i) {
            const float p = __expf(sc[i] - m_new);
            psum += p;
            pa[i] = (__bf16)p;
        }
        psum += __shfl_xor(psum, 16, 32);
        l = l * scale + psum;
        m = m_new;

        #pragma unroll
        for (int j = 0; j < 8; ++j) {
            const float sj = __shfl(scale, j + half * 8, 32);
            acc0[j] *= sj; acc1[j] *= sj; acc2[j] *= sj; acc3[j] *= sj;
        }

        const __bf16* p0 = hTb + (size_t)rl * N_ + c0 + half * 16;
        const v16bf bf0 = *(const v16bf*)(p0);
        const v16bf bf1 = *(const v16bf*)(p0 + (size_t)16 * N_);
        const v16bf bf2 = *(const v16bf*)(p0 + (size_t)32 * N_);
        const v16bf bf3 = *(const v16bf*)(p0 + (size_t)48 * N_);

        acc0 = __builtin_amdgcn_wmma_f32_16x16x32_bf16(false, pa, false, bf0, (short)0, acc0, false, false);
        acc1 = __builtin_amdgcn_wmma_f32_16x16x32_bf16(false, pa, false, bf1, (short)0, acc1, false, false);
        acc2 = __builtin_amdgcn_wmma_f32_16x16x32_bf16(false, pa, false, bf2, (short)0, acc2, false, false);
        acc3 = __builtin_amdgcn_wmma_f32_16x16x32_bf16(false, pa, false, bf3, (short)0, acc3, false, false);
    }

    // all TDM traffic + LDS reads done before overlaying accL onto adjbuf
    __syncthreads();

    #pragma unroll
    for (int j = 0; j < 8; ++j) {
        const int rj = j + half * 8;
        smem.accL[wv][rj][rl]      = acc0[j];
        smem.accL[wv][rj][rl + 16] = acc1[j];
        smem.accL[wv][rj][rl + 32] = acc2[j];
        smem.accL[wv][rj][rl + 48] = acc3[j];
    }
    if (lane < 16) { mL[wv][lane] = m; lL[wv][lane] = l; }
    __syncthreads();

    if (tid < 16) {
        float M = mL[0][tid];
        #pragma unroll
        for (int w = 1; w < 8; ++w) M = fmaxf(M, mL[w][tid]);
        float L = 0.f;
        #pragma unroll
        for (int w = 0; w < 8; ++w) {
            const float f = __expf(mL[w][tid] - M);
            wfac[w][tid] = f;
            L += f * lL[w][tid];
        }
        invL[tid] = 1.0f / L;
    }
    __syncthreads();

    const int col = tid & 63;
    const int rg  = tid >> 6;
    #pragma unroll
    for (int rr = 0; rr < 4; ++rr) {
        const int rowj = rg * 4 + rr;
        float sum = 0.f;
        #pragma unroll
        for (int w = 0; w < 8; ++w) sum += wfac[w][rowj] * smem.accL[w][rowj][col];
        float v = sum * invL[rowj];
        v = (v > 0.f) ? v : (__expf(v) - 1.0f);              // ELU
        __builtin_nontemporal_store(v, &out[((size_t)b * N_ + r0 + rowj) * FOUT_ + col]);
    }
}

// -------------------------------------------------------------------------
// Kernel 1: hidden = h @ W via fp32 WMMA 16x16x4.
// Fuses s1 = hidden.a1, s2 = hidden.a2 and stores hidden TRANSPOSED (bf16).
// W is staged in LDS K-paired so each B fragment is one ds_load_b64.
// -------------------------------------------------------------------------
__global__ __launch_bounds__(256) void gat_hidden_kernel(
    const float* __restrict__ h, const float* __restrict__ W,
    const float* __restrict__ aw, __bf16* __restrict__ hiddenT,
    float* __restrict__ s1, float* __restrict__ s2)
{
    // layout: Wl[(k>>1)*128 + n*2 + (k&1)]  -> (W[k][n], W[k+1][n]) contiguous
    __shared__ __align__(16) float Wl[FIN_ * FOUT_];   // 32 KB
    const int tid = threadIdx.x;
    {
        const float4* src = (const float4*)W;
        #pragma unroll
        for (int i = 0; i < (FIN_ * FOUT_ / 4) / 256; ++i) {
            const int idx4 = tid + i * 256;
            const float4 v = src[idx4];
            const int f0 = idx4 * 4;
            #pragma unroll
            for (int e = 0; e < 4; ++e) {
                const int f = f0 + e;
                const int k = f >> 6, n = f & 63;
                Wl[(k >> 1) * 128 + n * 2 + (k & 1)] =
                    (e == 0) ? v.x : (e == 1) ? v.y : (e == 2) ? v.z : v.w;
            }
        }
    }
    __syncthreads();

    const int lane = tid & 31, wave = tid >> 5;
    const int half = lane >> 4, rl = lane & 15;
    const int growBase = blockIdx.x * 128 + wave * 16;
    const int arow = growBase + rl;
    const int kk = half * 2;

    v8f acc0 = {}, acc1 = {}, acc2 = {}, acc3 = {};

    for (int k0 = 0; k0 < FIN_; k0 += 4) {
        v2f a;
        const float2 hv = *(const float2*)(h + (size_t)arow * FIN_ + k0 + kk);
        a[0] = hv.x; a[1] = hv.y;

        const float* wp = Wl + ((k0 + kk) >> 1) * 128;
        const v2f b0 = *(const v2f*)(wp + rl * 2);
        const v2f b1 = *(const v2f*)(wp + (rl + 16) * 2);
        const v2f b2 = *(const v2f*)(wp + (rl + 32) * 2);
        const v2f b3 = *(const v2f*)(wp + (rl + 48) * 2);

        acc0 = __builtin_amdgcn_wmma_f32_16x16x4_f32(false, a, false, b0, (short)0, acc0, false, false);
        acc1 = __builtin_amdgcn_wmma_f32_16x16x4_f32(false, a, false, b1, (short)0, acc1, false, false);
        acc2 = __builtin_amdgcn_wmma_f32_16x16x4_f32(false, a, false, b2, (short)0, acc2, false, false);
        acc3 = __builtin_amdgcn_wmma_f32_16x16x4_f32(false, a, false, b3, (short)0, acc3, false, false);
    }

    const float a10 = aw[rl],      a11 = aw[rl + 16], a12 = aw[rl + 32], a13 = aw[rl + 48];
    const float a20 = aw[64 + rl], a21 = aw[80 + rl], a22 = aw[96 + rl], a23 = aw[112 + rl];

    #pragma unroll
    for (int j = 0; j < 8; ++j) {
        const int grow = growBase + j + half * 8;
        const int bb = grow >> 11;
        const int rin = grow & (N_ - 1);
        __bf16* hTb = hiddenT + (size_t)bb * FOUT_ * N_;
        hTb[(size_t)(rl)      * N_ + rin] = (__bf16)acc0[j];
        hTb[(size_t)(rl + 16) * N_ + rin] = (__bf16)acc1[j];
        hTb[(size_t)(rl + 32) * N_ + rin] = (__bf16)acc2[j];
        hTb[(size_t)(rl + 48) * N_ + rin] = (__bf16)acc3[j];

        float p1 = acc0[j] * a10 + acc1[j] * a11 + acc2[j] * a12 + acc3[j] * a13;
        float p2 = acc0[j] * a20 + acc1[j] * a21 + acc2[j] * a22 + acc3[j] * a23;
        #pragma unroll
        for (int s = 1; s < 16; s <<= 1) {
            p1 += __shfl_xor(p1, s, 32);
            p2 += __shfl_xor(p2, s, 32);
        }
        if (rl == 0) {
            s1[bb * N_ + rin] = p1;
            s2[bb * N_ + rin] = p2;
        }
    }
}

// -------------------------------------------------------------------------
extern "C" void kernel_launch(void* const* d_in, const int* in_sizes, int n_in,
                              void* d_out, int out_size, void* d_ws, size_t ws_size,
                              hipStream_t stream)
{
    const float* h   = (const float*)d_in[0];   // (8, 2048, 128)
    const float* adj = (const float*)d_in[1];   // (8, 2048, 2048)
    const float* W   = (const float*)d_in[2];   // (128, 64)
    const float* aw  = (const float*)d_in[3];   // (128, 1)
    float* out = (float*)d_out;                 // (8, 2048, 64)

    char* ws = (char*)d_ws;
    __bf16* hiddenT = (__bf16*)ws;                                  // 2 MB
    float*  s1 = (float*)(ws + (size_t)B_ * FOUT_ * N_ * sizeof(__bf16));
    float*  s2 = s1 + B_ * N_;

    gat_hidden_kernel<<<128, 256, 0, stream>>>(h, W, aw, hiddenT, s1, s2);
    gat_attn_kernel<<<B_ * (N_ / 16), 256, 0, stream>>>(adj, hiddenT, s1, s2, out);
}
